// SimpleRetention_64536178590378
// MI455X (gfx1250) — matrix operand order repaired and checked by reading, not compiled, not run on hardware
//
#include <hip/hip_runtime.h>
#include <hip/hip_bf16.h>

// ---------------------------------------------------------------------------
// SimpleRetention fused bf16-WMMA implementation for gfx1250 (MI455X, wave32)
//   out[b,s,v] = sum_t ( (xpos(Q Wq)[b,s,:] . xpos_dn(K Wk)[b,t,:]) * D[s,t] )
//                * (V Wv)[b,t,v]
// ---------------------------------------------------------------------------

typedef __bf16 v16bf __attribute__((ext_vector_type(16)));
typedef __bf16 v8bf  __attribute__((ext_vector_type(8)));
typedef float  v8f   __attribute__((ext_vector_type(8)));

#define NB   4      // batch
#define SEQ  2048   // sequence length
#define HID  1024   // hidden
#define HD   512    // head dim

// log2(0.96875)
constexpr float LOG2_GAMMA = -0.045803415f;
// log2(10000)/256  (inv_freq exponent step, half = HD/2 = 256)
constexpr float L2_1E4_OVER_256 = 13.287712379549449f / 256.0f;

union BF16x16 { v16bf v; v8bf h[2]; };

__device__ __forceinline__ v8f vzero8() {
  v8f z;
#pragma unroll
  for (int i = 0; i < 8; ++i) z[i] = 0.0f;
  return z;
}

__device__ __forceinline__ v8f wmma_bf16(v16bf a, v16bf b, v8f c) {
  // D = A(16x32 bf16) * B(32x16 bf16) + C(16x16 f32)
  return __builtin_amdgcn_wmma_f32_16x16x32_bf16(
      /*neg_a=*/false, a, /*neg_b=*/false, b,
      /*c_mod=*/(short)0, c, /*reuse_a=*/false, /*reuse_b=*/false);
}

// exchange with lane^1 (wave32, group-of-32 ds_swizzle, SWAPX1: xor=1,and=0x1f)
__device__ __forceinline__ float lane_swap1(float x) {
  return __int_as_float(__builtin_amdgcn_ds_swizzle(__float_as_int(x), 0x041F));
}

// gfx1250 async global->LDS copy (ASYNCcnt-tracked; bypasses VGPRs).
// lds_off = byte offset into LDS (low 32 bits of generic shared pointer).
__device__ __forceinline__ void async_ld_b128(const void* g, unsigned lds_off) {
  asm volatile("global_load_async_to_lds_b128 %0, %1, off"
               :: "v"(lds_off), "v"(g)
               : "memory");
}
__device__ __forceinline__ void wait_async_zero() {
  asm volatile("s_wait_asynccnt 0x0" ::: "memory");
}

// ---------------------------------------------------------------------------
// Kernel 1: pre-swizzle the f32 weights into bf16 B-fragment layout.
// B-fragment convention (32x16 bf16 B-matrix): lane = (k_in/16)*16 + n_in,
// element e = k_in%16  ->  16 contiguous bf16 (32B) per lane per fragment.
// Wsw[((kstep*32 + ntile)*32 + lane)*16 + e]
// ---------------------------------------------------------------------------
__global__ __launch_bounds__(256) void k_swizzle_w(
    const float* __restrict__ Wq, const float* __restrict__ Wk,
    const float* __restrict__ Wv, __bf16* __restrict__ Wsw) {
  const float* W = (blockIdx.y == 0) ? Wq : (blockIdx.y == 1) ? Wk : Wv;
  __bf16* dst = Wsw + (size_t)blockIdx.y * (HID * HD);
  int idx = blockIdx.x * 256 + threadIdx.x;     // over HID*HD
  int k = idx >> 9, n = idx & 511;              // W is (1024,512) row-major
  int kstep = k >> 5, kin = k & 31, nt = n >> 4, nin = n & 15;
  int lane = ((kin >> 4) << 4) | nin;
  int e = kin & 15;
  dst[(size_t)((kstep * 32 + nt) * 32 + lane) * 16 + e] = (__bf16)W[idx];
}

// ---------------------------------------------------------------------------
// Kernel 2: projections X @ W (K=1024) via WMMA bf16, + xpos epilogue.
// grid = (SEQ/16, NB, 3mat), block = 256 (8 waves, wave w owns N-tiles w*4..w*4+3)
// ---------------------------------------------------------------------------
__global__ __launch_bounds__(256) void k_project(
    const float* __restrict__ Qin, const float* __restrict__ Kin,
    const float* __restrict__ Vin, const __bf16* __restrict__ Wsw,
    __bf16* __restrict__ Qbf, __bf16* __restrict__ Kbf,
    __bf16* __restrict__ Vsw) {
  __shared__ __align__(16) __bf16 As[16 * 1032];   // +16B stagger per row

  const int mat = blockIdx.z;
  const int b   = blockIdx.y;
  const int s0  = blockIdx.x * 16;
  const float* X = (mat == 0) ? Qin : (mat == 1) ? Kin : Vin;
  const __bf16* Wm = Wsw + (size_t)mat * (HID * HD);
  const int tid = threadIdx.x;

  // stage 16 x 1024 f32 panel -> bf16 LDS (conversion requires VGPR path)
  for (int idx = tid; idx < 16 * HID; idx += 256) {
    int r = idx >> 10, c = idx & (HID - 1);
    As[r * 1032 + c] = (__bf16)X[(size_t)(b * SEQ + s0 + r) * HID + c];
  }
  __syncthreads();

  const int lane = tid & 31, wave = tid >> 5;
  const int nin = lane & 15, halfsel = lane >> 4;

  v8f acc[4];
#pragma unroll
  for (int j = 0; j < 4; ++j) acc[j] = vzero8();

  for (int h = 0; h < HID / 32; ++h) {
    // A fragment (16x32): row m = lane%16, k = halfsel*16 + e (contiguous)
    BF16x16 a;
    const __bf16* ap = &As[nin * 1032 + h * 32 + halfsel * 16];
    a.h[0] = *(const v8bf*)ap;
    a.h[1] = *(const v8bf*)(ap + 8);
    // preload all 4 B fragments, then issue the 4 WMMAs (staggered waits)
    BF16x16 bb[4];
#pragma unroll
    for (int j = 0; j < 4; ++j) {
      const __bf16* bp =
          Wm + (size_t)((h * 32 + wave * 4 + j) * 32 + lane) * 16;
      bb[j].h[0] = *(const v8bf*)bp;
      bb[j].h[1] = *(const v8bf*)(bp + 8);
    }
#pragma unroll
    for (int j = 0; j < 4; ++j) acc[j] = wmma_bf16(a.v, bb[j].v, acc[j]);
  }

  if (mat < 2) {
    // xpos epilogue: out = x*cos*scale +/- pair*sin*scale ; K uses 1/scale
    const float sgn = (mat == 0) ? 1.0f : -1.0f;
    __bf16* Out = (mat == 0) ? Qbf : Kbf;
#pragma unroll
    for (int j = 0; j < 4; ++j) {
      int ng  = (wave * 4 + j) * 16 + nin;     // output dim 0..511
      int i2  = ng >> 1;                       // rotary pair index
      int par = ng & 1;
      float invf = exp2f(-(float)i2 * L2_1E4_OVER_256);
      // scale_vec = (2i + 0.4*512)/(1.4*512); exponent pos/512 (signed)
      float l2sv = log2f((2.0f * (float)i2 + 204.8f) * (1.0f / 716.8f)) *
                   sgn * (1.0f / 512.0f);
#pragma unroll
      for (int r = 0; r < 8; ++r) {
        int m = r + 8 * halfsel;               // C/D layout row
        float pos = (float)(s0 + m);
        float th = pos * invf;
        float sc = exp2f(l2sv * pos);
        float c2 = cosf(th) * sc;
        float s2 = sinf(th) * sc;
        float v = acc[j][r];
        float p = lane_swap1(v);               // partner dim (ng ^ 1)
        float res = par ? fmaf(v, c2, p * s2) : fmaf(v, c2, -p * s2);
        Out[(size_t)(b * SEQ + s0 + m) * HD + ng] = (__bf16)res;
      }
    }
  } else {
    // V: store directly in B-fragment swizzled layout (k_in = t % 32)
#pragma unroll
    for (int j = 0; j < 4; ++j) {
      int nt = wave * 4 + j;
#pragma unroll
      for (int r = 0; r < 8; ++r) {
        int t = s0 + r + 8 * halfsel;
        int kin = t & 31, tstep = t >> 5;
        int dl = ((kin >> 4) << 4) | nin;
        int e = kin & 15;
        Vsw[(size_t)b * (SEQ * HD) +
            (size_t)((tstep * 32 + nt) * 32 + dl) * 16 + e] = (__bf16)acc[j][r];
      }
    }
  }
}

// ---------------------------------------------------------------------------
// Kernel 3: fused retention. grid = (SEQ/16, NB), block = 256 (8 waves).
// Per workgroup: 16 query rows, stream causal 128-key blocks.
//   Phase A: wave w -> 16x16 score tile (QK^T over HD=512), apply gamma decay,
//            bf16 -> LDS (double-buffered -> single barrier per block).
//   Phase B: wave w -> 4 output N-tiles accumulate scores @ V (pre-swizzled).
// ---------------------------------------------------------------------------
__global__ __launch_bounds__(256) void k_retention(
    const __bf16* __restrict__ Qbf, const __bf16* __restrict__ Kbf,
    const __bf16* __restrict__ Vsw, float* __restrict__ out) {
  __shared__ __align__(16) __bf16 Qs[16 * 520];      // 16 x 512 + stagger
  __shared__ __align__(16) __bf16 Ss[2][16 * 136];   // double-buffered scores

  const int b  = blockIdx.y;
  const int s0 = blockIdx.x * 16;
  const int tid = threadIdx.x;

  // ---- Q panel: async global->LDS (ASYNCcnt), 4 x b128 per thread ----
#pragma unroll
  for (int c = 0; c < 4; ++c) {
    int chunk = tid + c * 256;                   // 1024 chunks of 8 bf16
    int r = chunk >> 6, col8 = (chunk & 63) << 3;
    const __bf16* g = Qbf + (size_t)(b * SEQ + s0 + r) * HD + col8;
    unsigned lds = (unsigned)(uintptr_t)(&Qs[r * 520 + col8]);
    async_ld_b128(g, lds);
  }
  wait_async_zero();
  __syncthreads();

  const int lane = tid & 31, wave = tid >> 5;
  const int nin = lane & 15, halfsel = lane >> 4;
  const size_t vbase = (size_t)b * (SEQ * HD);

  v8f acc[4];
#pragma unroll
  for (int j = 0; j < 4; ++j) acc[j] = vzero8();

  const int kbmax = s0 >> 7;                     // causal: key blocks of 128
  for (int kb = 0; kb <= kbmax; ++kb) {
    __bf16* Sb = Ss[kb & 1];
    // ---- Phase A: scores for key tile tb..tb+15 (this wave) ----
    const int tb = kb * 128 + wave * 16;
    const __bf16* krow = Kbf + (size_t)(b * SEQ + tb + nin) * HD;
    if (kb < kbmax) {
      // prefetch next key block (K rows for this wave, V fragments)
      __builtin_prefetch(krow + (size_t)128 * HD, 0, 0);
      __builtin_prefetch(krow + (size_t)128 * HD + 256, 0, 0);
      __builtin_prefetch(Vsw + vbase +
                             (size_t)(((kb + 1) * 4) * 32 + wave * 4) * 512 +
                             lane * 16,
                         0, 0);
    }
    v8f sacc = vzero8();
    for (int h = 0; h < HD / 32; ++h) {
      BF16x16 a;
      const __bf16* ap = &Qs[nin * 520 + h * 32 + halfsel * 16];
      a.h[0] = *(const v8bf*)ap;
      a.h[1] = *(const v8bf*)(ap + 8);
      // B = K^T fragment: lane col = key nin, k = halfsel*16 + e (contiguous)
      const __bf16* kp = krow + h * 32 + halfsel * 16;
      BF16x16 bb;
      bb.h[0] = *(const v8bf*)kp;
      bb.h[1] = *(const v8bf*)(kp + 8);
      sacc = wmma_bf16(a.v, bb.v, sacc);
    }
    // decay mask + bf16 -> LDS score panel (16 x 128)
#pragma unroll
    for (int r = 0; r < 8; ++r) {
      int m  = r + 8 * halfsel;
      int sg = s0 + m;
      int tg = tb + nin;
      float d = (sg >= tg) ? exp2f((float)(sg - tg) * LOG2_GAMMA) : 0.0f;
      Sb[m * 136 + wave * 16 + nin] = (__bf16)(sacc[r] * d);
    }
    __syncthreads();   // A-writes visible; double-buffer covers WAR on reuse

    // ---- Phase B: out += S(16x128) @ V(128 x 64 cols for this wave) ----
    for (int kk = 0; kk < 4; ++kk) {
      BF16x16 a;
      const __bf16* ap = &Sb[nin * 136 + kk * 32 + halfsel * 16];
      a.h[0] = *(const v8bf*)ap;
      a.h[1] = *(const v8bf*)(ap + 8);
      int tstep = kb * 4 + kk;
      BF16x16 bb[4];
#pragma unroll
      for (int j = 0; j < 4; ++j) {
        const __bf16* vp =
            Vsw + vbase + (size_t)((tstep * 32 + wave * 4 + j) * 32 + lane) * 16;
        bb[j].h[0] = *(const v8bf*)vp;
        bb[j].h[1] = *(const v8bf*)(vp + 8);
      }
#pragma unroll
      for (int j = 0; j < 4; ++j) acc[j] = wmma_bf16(a.v, bb[j].v, acc[j]);
    }
  }

  // f32 output, row-major (B,S,HD)
#pragma unroll
  for (int j = 0; j < 4; ++j) {
    int ng = (wave * 4 + j) * 16 + nin;
#pragma unroll
    for (int r = 0; r < 8; ++r) {
      int m = r + 8 * halfsel;
      out[(size_t)(b * SEQ + s0 + m) * HD + ng] = acc[j][r];
    }
  }
}

// ---------------------------------------------------------------------------
extern "C" void kernel_launch(void* const* d_in, const int* in_sizes, int n_in,
                              void* d_out, int out_size, void* d_ws,
                              size_t ws_size, hipStream_t stream) {
  (void)in_sizes; (void)n_in; (void)out_size; (void)ws_size;
  const float* Q  = (const float*)d_in[0];
  const float* K  = (const float*)d_in[1];
  const float* V  = (const float*)d_in[2];
  const float* Wq = (const float*)d_in[3];
  const float* Wk = (const float*)d_in[4];
  const float* Wv = (const float*)d_in[5];
  float* out = (float*)d_out;

  char* ws = (char*)d_ws;
  const size_t szP = (size_t)NB * SEQ * HD * 2;        // 8 MB bf16 panels
  __bf16* Qbf = (__bf16*)(ws);
  __bf16* Kbf = (__bf16*)(ws + szP);
  __bf16* Vsw = (__bf16*)(ws + 2 * szP);
  __bf16* Wsw = (__bf16*)(ws + 3 * szP);               // 3 x 1 MB

  dim3 blk(256);
  k_swizzle_w<<<dim3((HID * HD) / 256, 3), blk, 0, stream>>>(Wq, Wk, Wv, Wsw);
  k_project<<<dim3(SEQ / 16, NB, 3), blk, 0, stream>>>(Q, K, V, Wsw,
                                                       Qbf, Kbf, Vsw);
  k_retention<<<dim3(SEQ / 16, NB), blk, 0, stream>>>(Qbf, Kbf, Vsw, out);
}
